// FlashMQAttention_11656541241939
// MI455X (gfx1250) — matrix operand order repaired and checked by reading, not compiled, not run on hardware
//
#include <hip/hip_runtime.h>
#include <hip/hip_bf16.h>

// ---------------------------------------------------------------------------
// CDNA5 (gfx1250) fused MQA block, bf16 WMMA pipeline:
//   pass0: fp32 -> bf16 conversion (activations) and convert+transpose (weights)
//   pass1: qkv  = hidden @ w_attn + b_attn          (bf16 in, bf16 out)
//   pass2: causal MQA flash attention               (bf16 in, bf16 out)
//   pass3: out  = attn @ w_proj + b_proj            (bf16 in, fp32 out)
// Matrix math: v_wmma_f32_16x16x32_bf16.
// Staging: global_load_async_to_lds_b128 with double-buffered (ping-pong) LDS
// in the GEMMs: tile i+1 streams in behind the WMMAs on tile i, drained with
// s_wait_asynccnt 0x8 (older 8 async ops complete, newest 8 stay in flight).
// ---------------------------------------------------------------------------

typedef __attribute__((ext_vector_type(16))) __bf16          v16bf;
typedef __attribute__((ext_vector_type(8)))  float           v8f;
typedef __attribute__((ext_vector_type(8)))  unsigned short  ushort8;
typedef __attribute__((ext_vector_type(4)))  unsigned short  ushort4_t;

// fp32 -> bf16 round-to-nearest-even
__device__ __forceinline__ unsigned short f2bf(float f) {
    union { float f; unsigned int u; } v; v.f = f;
    unsigned int u = v.u;
    unsigned int r = u + 0x7FFFu + ((u >> 16) & 1u);
    return (unsigned short)(r >> 16);
}

// CDNA5 async global->LDS copy, 16 bytes per lane. LDS offset = low 32 bits of
// the generic pointer; tracked by ASYNCcnt.
__device__ __forceinline__ void async_copy_b128(const unsigned short* gsrc,
                                                unsigned short* ldst) {
    unsigned int       lds = (unsigned int)(unsigned long long)ldst;
    unsigned long long ga  = (unsigned long long)gsrc;
    asm volatile("global_load_async_to_lds_b128 %0, %1, off"
                 :: "v"(lds), "v"(ga) : "memory");
}
__device__ __forceinline__ void async_wait_all() {
    asm volatile("s_wait_asynccnt 0x0" ::: "memory");
}
// Wait until only the newest 8 async ops remain outstanding (per-wave).
__device__ __forceinline__ void async_wait_keep8() {
    asm volatile("s_wait_asynccnt 0x8" ::: "memory");
}

// A-fragment (16x32 bf16, MxK): lane m = lane&15, K-half kh = lane>>4.
// elems 0..7 -> K = kh*8 + 0..7 ; elems 8..15 -> K = 16 + kh*8 + 0..7.
__device__ __forceinline__ v16bf load_a_frag(const unsigned short* lds, int ld,
                                             int mbase, int kbase) {
    int lane = threadIdx.x & 31;
    int m  = lane & 15;
    int kh = lane >> 4;
    const unsigned short* p = lds + (mbase + m) * ld + kbase + kh * 8;
    union { ushort8 h[2]; v16bf v; } u;
    u.h[0] = *(const ushort8*)(p);
    u.h[1] = *(const ushort8*)(p + 16);
    return u.v;
}

// B-fragment (32x16 bf16, KxN): lane n = lane&15, K-half kh = lane>>4.
// elems 0..7 -> K = kh*8 + 0..7 ; elems 8..15 -> K = 16 + kh*8 + 0..7. LDS tile stored transposed: [n][k].
__device__ __forceinline__ v16bf load_b_frag(const unsigned short* lds, int ld,
                                             int nbase, int kbase) {
    int lane = threadIdx.x & 31;
    int n  = lane & 15;
    int kh = lane >> 4;
    const unsigned short* p = lds + (nbase + n) * ld + kbase + kh * 8;
    union { ushort8 h[2]; v16bf v; } u;
    u.h[0] = *(const ushort8*)(p);
    u.h[1] = *(const ushort8*)(p + 16);
    return u.v;
}

__device__ __forceinline__ v8f wmma_bf16(v16bf a, v16bf b, v8f c) {
    return __builtin_amdgcn_wmma_f32_16x16x32_bf16(
        false, a, false, b, (short)0, c, false, false);
}

// ---------------------------------------------------------------------------
// Pass 0a: flat fp32 -> bf16 (n multiple of 4)
// ---------------------------------------------------------------------------
__global__ __launch_bounds__(256)
void conv_bf16_kernel(const float* __restrict__ src, unsigned short* __restrict__ dst,
                      int n) {
    int i = (blockIdx.x * 256 + threadIdx.x) * 4;
    if (i >= n) return;
    const float4 s = *(const float4*)(src + i);
    ushort4_t o;
    o.x = f2bf(s.x); o.y = f2bf(s.y); o.z = f2bf(s.z); o.w = f2bf(s.w);
    *(ushort4_t*)(dst + i) = o;
}

// ---------------------------------------------------------------------------
// Pass 0b: weight convert + transpose: src[k][n] fp32 -> dst[n][k] bf16.
// Block handles a 32(n) x 8(k) patch; reads coalesced along n.
// ---------------------------------------------------------------------------
__global__ __launch_bounds__(256)
void conv_transpose_kernel(const float* __restrict__ src, unsigned short* __restrict__ dst,
                           int K, int N) {
    int n = blockIdx.x * 32 + (threadIdx.x & 31);
    int k = blockIdx.y * 8 + (threadIdx.x >> 5);
    dst[(size_t)n * K + k] = f2bf(src[(size_t)k * N + n]);
}

// ---------------------------------------------------------------------------
// GEMM: C = A(MxK,bf16) * WT(NxK,bf16 transposed) + bias(fp32)
// C written as bf16 (out_bf16=1) or fp32. M,N mult of 128; K mult of 64.
// 256 threads = 8 waves (2x4 grid), 128x128 C tile, BK=64, double-buffered LDS.
// ---------------------------------------------------------------------------
#define GEMM_LD 72   // 64 + 8 pad ushorts; 144B rows stay 16B aligned

// Issue the 8 per-thread async copies staging one (A,B) K-tile pair.
__device__ __forceinline__ void gemm_stage(const unsigned short* __restrict__ A,
                                           const unsigned short* __restrict__ WT,
                                           int m0, int n0, int K, int k0,
                                           unsigned short* As, unsigned short* Bs,
                                           int tid) {
    #pragma unroll
    for (int it = 0; it < 4; ++it) {
        int c  = tid + it * 256;
        int r  = c >> 3;
        int ch = (c & 7) * 8;
        async_copy_b128(A + (size_t)(m0 + r) * K + k0 + ch, As + r * GEMM_LD + ch);
    }
    #pragma unroll
    for (int it = 0; it < 4; ++it) {
        int c  = tid + it * 256;
        int r  = c >> 3;
        int ch = (c & 7) * 8;
        async_copy_b128(WT + (size_t)(n0 + r) * K + k0 + ch, Bs + r * GEMM_LD + ch);
    }
}

__global__ __launch_bounds__(256)
void gemm_bf16_kernel(const unsigned short* __restrict__ A,
                      const unsigned short* __restrict__ WT,
                      const float* __restrict__ bias, void* __restrict__ Cout,
                      int M, int N, int K, int out_bf16) {
    __shared__ unsigned short Asm[2][128 * GEMM_LD];
    __shared__ unsigned short Bsm[2][128 * GEMM_LD];

    const int tid  = threadIdx.x;
    const int m0   = blockIdx.y * 128;
    const int n0   = blockIdx.x * 128;
    const int wid  = tid >> 5;
    const int wm   = wid >> 2;       // 0..1  -> 64 rows
    const int wn   = wid & 3;        // 0..3  -> 32 cols
    const int lane = tid & 31;

    v8f acc[4][2];
    #pragma unroll
    for (int mt = 0; mt < 4; ++mt)
        #pragma unroll
        for (int nt = 0; nt < 2; ++nt)
            #pragma unroll
            for (int e = 0; e < 8; ++e) acc[mt][nt][e] = 0.0f;

    const int nk = K >> 6;
    gemm_stage(A, WT, m0, n0, K, 0, Asm[0], Bsm[0], tid);   // prologue: tile 0

    for (int i = 0; i < nk; ++i) {
        const int cur = i & 1;
        __syncthreads();   // all waves done reading buf[cur^1] before it is rewritten
        if (i + 1 < nk) {
            gemm_stage(A, WT, m0, n0, K, (i + 1) << 6, Asm[cur ^ 1], Bsm[cur ^ 1], tid);
            async_wait_keep8();   // tile i complete; tile i+1 stays in flight
        } else {
            async_wait_all();
        }
        __syncthreads();

        #pragma unroll
        for (int ks = 0; ks < 64; ks += 32) {
            v16bf af[4], bfr[2];
            #pragma unroll
            for (int mt = 0; mt < 4; ++mt)
                af[mt] = load_a_frag(Asm[cur], GEMM_LD, wm * 64 + mt * 16, ks);
            #pragma unroll
            for (int nt = 0; nt < 2; ++nt)
                bfr[nt] = load_b_frag(Bsm[cur], GEMM_LD, wn * 32 + nt * 16, ks);
            #pragma unroll
            for (int mt = 0; mt < 4; ++mt)
                #pragma unroll
                for (int nt = 0; nt < 2; ++nt)
                    acc[mt][nt] = wmma_bf16(af[mt], bfr[nt], acc[mt][nt]);
        }
    }

    const int cn    = lane & 15;
    const int rbase = 8 * (lane >> 4);
    #pragma unroll
    for (int mt = 0; mt < 4; ++mt) {
        #pragma unroll
        for (int nt = 0; nt < 2; ++nt) {
            int col  = n0 + wn * 32 + nt * 16 + cn;
            float bv = bias[col];
            #pragma unroll
            for (int v = 0; v < 8; ++v) {
                int row = m0 + wm * 64 + mt * 16 + rbase + v;
                float val = acc[mt][nt][v] + bv;
                if (out_bf16)
                    ((unsigned short*)Cout)[(size_t)row * N + col] = f2bf(val);
                else
                    ((float*)Cout)[(size_t)row * N + col] = val;
            }
        }
    }
}

// ---------------------------------------------------------------------------
// Causal MQA flash attention on bf16 qkv [4096][2304]:
// Q = [:, h*128 : h*128+128], K = [:, 2048:2176], V = [:, 2176:2304].
// One block per (q-tile of 128 rows, head, batch); bf16 output.
// ---------------------------------------------------------------------------
#define ATT_LD 136
#define SLD    132

__global__ __launch_bounds__(256)
void mqa_attn_kernel(const unsigned short* __restrict__ qkv,
                     unsigned short* __restrict__ attn) {
    extern __shared__ char smem[];
    unsigned short* Qs = (unsigned short*)smem;        // [128][136]
    unsigned short* Ks = Qs + 128 * ATT_LD;            // [s][d]  (B^T for QK^T)
    unsigned short* Vs = Ks + 128 * ATT_LD;            // [d][s]  (transposed V)
    unsigned short* Ps = Vs + 128 * ATT_LD;            // [q][s] bf16 probs
    float* Sl        = (float*)(Ps + 128 * ATT_LD);    // [128][132] fp32 scores
    float* row_m     = Sl + 128 * SLD;
    float* row_l     = row_m + 128;
    float* row_scale = row_l + 128;

    const int tid  = threadIdx.x;
    const int qt   = blockIdx.x;
    const int h    = blockIdx.y;
    const int b    = blockIdx.z;
    const int wid  = tid >> 5;
    const int wm   = wid >> 2;
    const int wn   = wid & 3;
    const int lane = tid & 31;
    const float scale = 0.08838834764831845f;  // 128^-0.5

    const unsigned short* qbase = qkv + (size_t)(b * 1024 + qt * 128) * 2304 + h * 128;

    // Async-stage Q tile (128 x 128 bf16 = 2048 16B chunks), resident all iters.
    #pragma unroll
    for (int c = tid; c < 2048; c += 256) {
        int r = c >> 4, ch = (c & 15) * 8;
        async_copy_b128(qbase + (size_t)r * 2304 + ch, Qs + r * ATT_LD + ch);
    }
    if (tid < 128) { row_m[tid] = -3.0e38f; row_l[tid] = 0.0f; }

    v8f oacc[4][2];
    #pragma unroll
    for (int mt = 0; mt < 4; ++mt)
        #pragma unroll
        for (int nt = 0; nt < 2; ++nt)
            #pragma unroll
            for (int e = 0; e < 8; ++e) oacc[mt][nt][e] = 0.0f;
    async_wait_all();
    __syncthreads();

    for (int kt = 0; kt <= qt; ++kt) {
        const unsigned short* kbase = qkv + (size_t)(b * 1024 + kt * 128) * 2304 + 2048;
        const unsigned short* vbase = kbase + 128;
        // K tile: async row-major copy. V tile: b128 loads + transposed stores.
        #pragma unroll
        for (int c = tid; c < 2048; c += 256) {
            int r = c >> 4, ch = (c & 15) * 8;
            async_copy_b128(kbase + (size_t)r * 2304 + ch, Ks + r * ATT_LD + ch);
        }
        #pragma unroll
        for (int c = tid; c < 2048; c += 256) {
            int s = c >> 4, ch = (c & 15) * 8;
            ushort8 vv = *(const ushort8*)(vbase + (size_t)s * 2304 + ch);
            #pragma unroll
            for (int e = 0; e < 8; ++e)
                Vs[(ch + e) * ATT_LD + s] = vv[e];
        }
        async_wait_all();
        __syncthreads();

        // S = Q K^T
        v8f sacc[4][2];
        #pragma unroll
        for (int mt = 0; mt < 4; ++mt)
            #pragma unroll
            for (int nt = 0; nt < 2; ++nt)
                #pragma unroll
                for (int e = 0; e < 8; ++e) sacc[mt][nt][e] = 0.0f;
        #pragma unroll
        for (int k0 = 0; k0 < 128; k0 += 32) {
            v16bf af[4], bfr[2];
            #pragma unroll
            for (int mt = 0; mt < 4; ++mt)
                af[mt] = load_a_frag(Qs, ATT_LD, wm * 64 + mt * 16, k0);
            #pragma unroll
            for (int nt = 0; nt < 2; ++nt)
                bfr[nt] = load_b_frag(Ks, ATT_LD, wn * 32 + nt * 16, k0);
            #pragma unroll
            for (int mt = 0; mt < 4; ++mt)
                #pragma unroll
                for (int nt = 0; nt < 2; ++nt)
                    sacc[mt][nt] = wmma_bf16(af[mt], bfr[nt], sacc[mt][nt]);
        }
        #pragma unroll
        for (int mt = 0; mt < 4; ++mt)
            #pragma unroll
            for (int nt = 0; nt < 2; ++nt)
                #pragma unroll
                for (int v = 0; v < 8; ++v) {
                    int r = wm * 64 + mt * 16 + 8 * (lane >> 4) + v;
                    int c = wn * 32 + nt * 16 + (lane & 15);
                    Sl[r * SLD + c] = sacc[mt][nt][v];
                }
        __syncthreads();

        // Online softmax, one row per thread, float4 vectorized.
        if (tid < 128) {
            int r   = tid;
            int lim = (kt == qt) ? (r + 1) : 128;   // causal mask on diagonal tile
            const float4* srow4 = (const float4*)(Sl + r * SLD);
            float mmax = -3.0e38f;
            for (int c4 = 0; c4 < 32; ++c4) {
                float4 s = srow4[c4];
                int base = c4 * 4;
                if (base + 0 < lim) mmax = fmaxf(mmax, s.x * scale);
                if (base + 1 < lim) mmax = fmaxf(mmax, s.y * scale);
                if (base + 2 < lim) mmax = fmaxf(mmax, s.z * scale);
                if (base + 3 < lim) mmax = fmaxf(mmax, s.w * scale);
            }
            float mold = row_m[r];
            float mnew = fmaxf(mold, mmax);
            float sc   = __expf(mold - mnew);
            float lsum = 0.0f;
            ushort4_t* prow4 = (ushort4_t*)(Ps + r * ATT_LD);
            for (int c4 = 0; c4 < 32; ++c4) {
                float4 s = srow4[c4];
                int base = c4 * 4;
                float p0 = (base + 0 < lim) ? __expf(s.x * scale - mnew) : 0.0f;
                float p1 = (base + 1 < lim) ? __expf(s.y * scale - mnew) : 0.0f;
                float p2 = (base + 2 < lim) ? __expf(s.z * scale - mnew) : 0.0f;
                float p3 = (base + 3 < lim) ? __expf(s.w * scale - mnew) : 0.0f;
                ushort4_t o;
                o.x = f2bf(p0); o.y = f2bf(p1); o.z = f2bf(p2); o.w = f2bf(p3);
                prow4[c4] = o;
                lsum += (p0 + p1) + (p2 + p3);
            }
            row_m[r]     = mnew;
            row_l[r]     = row_l[r] * sc + lsum;
            row_scale[r] = sc;
        }
        __syncthreads();

        // Rescale running O, then O += P V
        #pragma unroll
        for (int mt = 0; mt < 4; ++mt)
            #pragma unroll
            for (int nt = 0; nt < 2; ++nt)
                #pragma unroll
                for (int v = 0; v < 8; ++v) {
                    int r = wm * 64 + mt * 16 + 8 * (lane >> 4) + v;
                    oacc[mt][nt][v] *= row_scale[r];
                }
        #pragma unroll
        for (int k0 = 0; k0 < 128; k0 += 32) {
            v16bf af[4], bfr[2];
            #pragma unroll
            for (int mt = 0; mt < 4; ++mt)
                af[mt] = load_a_frag(Ps, ATT_LD, wm * 64 + mt * 16, k0);
            #pragma unroll
            for (int nt = 0; nt < 2; ++nt)
                bfr[nt] = load_b_frag(Vs, ATT_LD, wn * 32 + nt * 16, k0);
            #pragma unroll
            for (int mt = 0; mt < 4; ++mt)
                #pragma unroll
                for (int nt = 0; nt < 2; ++nt)
                    oacc[mt][nt] = wmma_bf16(af[mt], bfr[nt], oacc[mt][nt]);
        }
        __syncthreads();
    }

    // Normalize, store bf16: attn[(b*1024+qt*128+r)][h*128+col]
    #pragma unroll
    for (int mt = 0; mt < 4; ++mt)
        #pragma unroll
        for (int nt = 0; nt < 2; ++nt) {
            int col = wn * 32 + nt * 16 + (lane & 15);
            #pragma unroll
            for (int v = 0; v < 8; ++v) {
                int r = wm * 64 + mt * 16 + 8 * (lane >> 4) + v;
                float invl = 1.0f / row_l[r];
                size_t row = (size_t)(b * 1024 + qt * 128 + r);
                attn[row * 2048 + h * 128 + col] = f2bf(oacc[mt][nt][v] * invl);
            }
        }
}

// ---------------------------------------------------------------------------
extern "C" void kernel_launch(void* const* d_in, const int* in_sizes, int n_in,
                              void* d_out, int out_size, void* d_ws, size_t ws_size,
                              hipStream_t stream) {
    const float* hidden = (const float*)d_in[0];  // [4096][2048]
    const float* w_attn = (const float*)d_in[1];  // [2048][2304]
    const float* b_attn = (const float*)d_in[2];  // [2304]
    const float* w_proj = (const float*)d_in[3];  // [2048][2048]
    const float* b_proj = (const float*)d_in[4];  // [2048]
    float* out = (float*)d_out;                   // [4096][2048] fp32

    const int T = 4096, HID = 2048, QKVN = 2304;

    // Workspace carve (all bf16 as ushort)
    unsigned short* hidden_bf = (unsigned short*)d_ws;              // T*HID
    unsigned short* wattnT    = hidden_bf + (size_t)T * HID;        // QKVN*HID ([n][k])
    unsigned short* wprojT    = wattnT + (size_t)QKVN * HID;        // HID*HID  ([n][k])
    unsigned short* qkv_bf    = wprojT + (size_t)HID * HID;         // T*QKVN
    unsigned short* attn_bf   = qkv_bf + (size_t)T * QKVN;          // T*HID

    // Pass 0: conversions
    conv_bf16_kernel<<<(T * HID) / 1024, 256, 0, stream>>>(hidden, hidden_bf, T * HID);
    conv_transpose_kernel<<<dim3(QKVN / 32, HID / 8), 256, 0, stream>>>(
        w_attn, wattnT, HID, QKVN);
    conv_transpose_kernel<<<dim3(HID / 32, HID / 8), 256, 0, stream>>>(
        w_proj, wprojT, HID, HID);

    // Pass 1: qkv = hidden @ w_attn + b_attn  (bf16 out)
    gemm_bf16_kernel<<<dim3(QKVN / 128, T / 128), 256, 0, stream>>>(
        hidden_bf, wattnT, b_attn, qkv_bf, T, QKVN, HID, 1);

    // Pass 2: causal MQA flash attention (bf16 out)
    size_t smem = (size_t)4 * 128 * ATT_LD * sizeof(unsigned short)
                + (size_t)128 * SLD * sizeof(float)
                + (size_t)3 * 128 * sizeof(float);
    mqa_attn_kernel<<<dim3(8, 16, 4), 256, smem, stream>>>(qkv_bf, attn_bf);

    // Pass 3: out = attn @ w_proj + b_proj (fp32 out)
    gemm_bf16_kernel<<<dim3(HID / 128, T / 128), 256, 0, stream>>>(
        attn_bf, wprojT, b_proj, out, T, HID, HID, 0);
}
